// AbsPosSelfAttention_13322988552734
// MI455X (gfx1250) — matrix operand order repaired and checked by reading, not compiled, not run on hardware
//
#include <hip/hip_runtime.h>
#include <hip/hip_bf16.h>

// AbsPosSelfAttention as flash-attention over L=4096, d=32, 16 (b,n) heads.
// logits = q·(k + emb_h[p] + emb_w[q])  ->  fold emb into K' at stage time.
// Transposed-S formulation: S^T = K' x Q^T so softmax state is per-lane and
// P converts C-layout -> A-operand by pure in-lane bf16 packing (cvt_pk).
// Row sums of P are computed on the matrix unit via an all-ones B operand.

typedef __attribute__((ext_vector_type(16))) __bf16 v16bf;
typedef __attribute__((ext_vector_type(2)))  __bf16 v2bf;
typedef __attribute__((ext_vector_type(8)))  float  v8f;

#define BS_      2
#define HEADS_   8
#define HH_      64
#define WW_      64
#define DIM_     32
#define L_       (HH_ * WW_)        // 4096
#define RPB_     128                // query rows per block (8 waves x 16)
#define KC_      64                 // keys per chunk (4 S-tiles)
#define NCHUNK_  (L_ / KC_)         // 64

// A-matrix (16x32 bf16) operand gather: lane row = rowbase entity (lane%16),
// VGPR v holds K = 2v + 8*half (v<4) / K = 2v + 8 + 8*half (v>=4).
__device__ __forceinline__ v16bf gatherA(const __bf16* rowbase, int half) {
  const unsigned* p = (const unsigned*)rowbase;
  union { unsigned u[8]; v16bf v; } r;
  const int h4 = half * 4;
#pragma unroll
  for (int j = 0; j < 4; ++j) r.u[j] = p[h4 + j];
#pragma unroll
  for (int j = 0; j < 4; ++j) r.u[4 + j] = p[h4 + 8 + j];
  return r.v;
}

// B-matrix (32x16 bf16) operand gather: lane col = rowbase entity (lane%16),
// VGPR v holds K = 2v + 8*half (v<4) / K = 2v + 8 + 8*half (v>=4).
__device__ __forceinline__ v16bf gatherB(const __bf16* rowbase, int half) {
  const unsigned* p = (const unsigned*)rowbase;
  union { unsigned u[8]; v16bf v; } r;
  const int h4 = half * 4;
#pragma unroll
  for (int j = 0; j < 4; ++j) r.u[j] = p[h4 + j];
#pragma unroll
  for (int j = 0; j < 4; ++j) r.u[4 + j] = p[h4 + 8 + j];
  return r.v;
}

// Single-instruction packed f32->bf16 pair convert (v_cvt_pk_bf16_f32).
__device__ __forceinline__ unsigned pk_bf16(float a, float b) {
#if __has_builtin(__builtin_amdgcn_cvt_pk_bf16_f32)
  v2bf t = __builtin_amdgcn_cvt_pk_bf16_f32(a, b);
  return __builtin_bit_cast(unsigned, t);
#else
  unsigned short x = __builtin_bit_cast(unsigned short, (__bf16)a);
  unsigned short y = __builtin_bit_cast(unsigned short, (__bf16)b);
  return (unsigned)x | ((unsigned)y << 16);
#endif
}

__device__ __forceinline__ v8f wmma_bf16(v16bf a, v16bf b, v8f c) {
  return __builtin_amdgcn_wmma_f32_16x16x32_bf16(false, a, false, b,
                                                 (short)0, c, false, false);
}

__global__ __launch_bounds__(256)
void abspos_attn_kernel(const float* __restrict__ Qg, const float* __restrict__ Kg,
                        const float* __restrict__ Vg, const float* __restrict__ Eh,
                        const float* __restrict__ Ew, float* __restrict__ Og) {
  __shared__ alignas(16) __bf16 Qs[RPB_ * DIM_];   // 8 KB: scaled Q [row][d]
  __shared__ alignas(16) __bf16 Ks[KC_ * DIM_];    // 4 KB: K' chunk [slot][d]
  __shared__ alignas(16) __bf16 Vt[DIM_ * KC_];    // 4 KB: V chunk  [d][slot]

  const int tid  = threadIdx.x;
  const int wave = tid >> 5;
  const int lane = tid & 31;
  const int l16  = lane & 15;
  const int half = (lane >> 4) & 1;

  const int tileIdx = blockIdx.x & 31;     // 32 row tiles per head
  const int bn      = blockIdx.x >> 5;     // 0..15  (b*8 + n)
  const int b       = bn >> 3;
  const int n       = bn & 7;
  const int row0    = tileIdx * RPB_;

  const size_t headBase = (size_t)bn * L_ * DIM_;

  // ---- stage Q tile (scaled by dim^-0.5) as bf16, vectorized ----
  for (int e = tid * 4; e < RPB_ * DIM_; e += 256 * 4) {
    const float4 qv = *(const float4*)(Qg + headBase + (size_t)row0 * DIM_ + e);
    const float s = 0.17677669529663687f;
    *(uint2*)&Qs[e] = uint2{pk_bf16(qv.x * s, qv.y * s), pk_bf16(qv.z * s, qv.w * s)};
  }
  __syncthreads();

  // Loop-invariant B operand: Q^T (d x qrow), col = this lane's query row.
  const v16bf qb = gatherB(&Qs[(wave * 16 + l16) * DIM_], half);

  // All-ones B operand for matrix-unit row sums (bf16 1.0 = 0x3F80).
  // Formally lane-dependent so it is kept in VGPRs (no per-use SGPR->VGPR movs).
  union { unsigned u[8]; v16bf v; } ones;
#pragma unroll
  for (int j = 0; j < 8; ++j) ones.u[j] = 0x3F803F80u | ((unsigned)tid & 0u);

  float m = -1e30f;                 // per-lane running max for qrow = l16
  v8f o0 = {}, o1 = {};             // O accumulators (row i+8h, col l16)
  v8f lacc = {};                    // row-sum accumulator, same layout

  // Staging: each thread owns a key PAIR (slot-adjacent) x 4-dim segment.
  const int kp = tid >> 3;          // 0..31 -> keys 2kp, 2kp+1
  const int sd = (tid & 7) * 4;     // 0..28

  for (int ck = 0; ck < NCHUNK_; ++ck) {
    __syncthreads();                // prior chunk's LDS reads done
    // ---- stage K' = k + emb_h[p] + emb_w[q] and V^T (bf16) ----
    {
      const int gk0 = ck * KC_ + kp * 2;       // even; pair shares emb_h row
      const int p   = gk0 >> 6;
      const int q0  = gk0 & 63;
      const size_t gidx = headBase + (size_t)gk0 * DIM_ + sd;
      const float4 k0 = *(const float4*)(Kg + gidx);
      const float4 k1 = *(const float4*)(Kg + gidx + DIM_);
      const float4 v0 = *(const float4*)(Vg + gidx);
      const float4 v1 = *(const float4*)(Vg + gidx + DIM_);
      const float4 eh = *(const float4*)(Eh + p * DIM_ + sd);
      const float4 e0 = *(const float4*)(Ew + q0 * DIM_ + sd);
      const float4 e1 = *(const float4*)(Ew + (q0 + 1) * DIM_ + sd);
      *(uint2*)&Ks[(kp * 2) * DIM_ + sd] =
          uint2{pk_bf16(k0.x + eh.x + e0.x, k0.y + eh.y + e0.y),
                pk_bf16(k0.z + eh.z + e0.z, k0.w + eh.w + e0.w)};
      *(uint2*)&Ks[(kp * 2 + 1) * DIM_ + sd] =
          uint2{pk_bf16(k1.x + eh.x + e1.x, k1.y + eh.y + e1.y),
                pk_bf16(k1.z + eh.z + e1.z, k1.w + eh.w + e1.w)};
      // V^T: pack slot-adjacent pair per dim -> 4 x b32 stores
      *(unsigned*)&Vt[(sd + 0) * KC_ + kp * 2] = pk_bf16(v0.x, v1.x);
      *(unsigned*)&Vt[(sd + 1) * KC_ + kp * 2] = pk_bf16(v0.y, v1.y);
      *(unsigned*)&Vt[(sd + 2) * KC_ + kp * 2] = pk_bf16(v0.z, v1.z);
      *(unsigned*)&Vt[(sd + 3) * KC_ + kp * 2] = pk_bf16(v0.w, v1.w);
      if (ck + 1 < NCHUNK_) {
        __builtin_prefetch(Kg + gidx + KC_ * DIM_, 0, 3);
        __builtin_prefetch(Vg + gidx + KC_ * DIM_, 0, 3);
      }
    }
    __syncthreads();

    // ---- S^T tiles: A = K' slots, B = Q^T (4 WMMAs, K=32) ----
    v8f s[4];
    v8f zc = {};
#pragma unroll
    for (int t = 0; t < 4; ++t) {
      const v16bf ka = gatherA(&Ks[(t * 16 + l16) * DIM_], half);
      s[t] = wmma_bf16(ka, qb, zc);   // lane: qrow=l16, slots t*16 + i+8h
    }

    // ---- online softmax: in-lane tree + one cross-half exchange ----
    float mloc = s[0][0];
#pragma unroll
    for (int t = 0; t < 4; ++t)
#pragma unroll
      for (int i = (t == 0); i < 8; ++i) mloc = fmaxf(mloc, s[t][i]);
    mloc = fmaxf(mloc, __shfl_xor(mloc, 16, 32));
    const bool  up   = mloc > m;
    const float mnew = fmaxf(m, mloc);
    const float sc   = __expf(m - mnew);
    m = mnew;

    // Rescale O and row-sum acc only when some row's max increased.
    if (__any(up)) {
#pragma unroll
      for (int i = 0; i < 8; ++i) {
        const float scr = __shfl(sc, i + 8 * half, 32);  // sc of row i+8h
        o0[i] *= scr;
        o1[i] *= scr;
        lacc[i] *= scr;
      }
    }

    // ---- P = exp(S - m): C-layout -> A-operand via fused cvt_pk packing ----
    union { unsigned u[8]; v16bf v; } pa0, pa1;
#pragma unroll
    for (int j = 0; j < 4; ++j) {
      pa0.u[j]     = pk_bf16(__expf(s[0][2*j] - mnew), __expf(s[0][2*j+1] - mnew));
      pa0.u[4 + j] = pk_bf16(__expf(s[1][2*j] - mnew), __expf(s[1][2*j+1] - mnew));
      pa1.u[j]     = pk_bf16(__expf(s[2][2*j] - mnew), __expf(s[2][2*j+1] - mnew));
      pa1.u[4 + j] = pk_bf16(__expf(s[3][2*j] - mnew), __expf(s[3][2*j+1] - mnew));
    }

    // ---- row sums on the matrix unit: lacc += P x ones ----
    lacc = wmma_bf16(pa0.v, ones.v, lacc);
    lacc = wmma_bf16(pa1.v, ones.v, lacc);

    // ---- O += P x V (4 WMMAs); B = V^T per (dim subtile, slot half) ----
    const v16bf vb00 = gatherB(&Vt[l16 * KC_], half);              // d 0-15, s 0-31
    const v16bf vb01 = gatherB(&Vt[l16 * KC_ + 32], half);         // d 0-15, s 32-63
    const v16bf vb10 = gatherB(&Vt[(16 + l16) * KC_], half);       // d 16-31, s 0-31
    const v16bf vb11 = gatherB(&Vt[(16 + l16) * KC_ + 32], half);  // d 16-31, s 32-63
    o0 = wmma_bf16(pa0.v, vb00, o0);
    o0 = wmma_bf16(pa1.v, vb01, o0);
    o1 = wmma_bf16(pa0.v, vb10, o1);
    o1 = wmma_bf16(pa1.v, vb11, o1);
  }

  // ---- normalize and store: out[b][row][n*32 + d]; O row = i+8h, col = l16.
  // lacc[i] already holds the row sum for row i+8h (replicated per column).
#pragma unroll
  for (int i = 0; i < 8; ++i) {
    const float inv = 1.0f / lacc[i];
    const int   row = row0 + wave * 16 + i + 8 * half;
    const size_t ob = (((size_t)b * L_ + row) * (HEADS_ * DIM_)) + (size_t)n * DIM_;
    Og[ob + l16]      = o0[i] * inv;
    Og[ob + 16 + l16] = o1[i] * inv;
  }
}

extern "C" void kernel_launch(void* const* d_in, const int* in_sizes, int n_in,
                              void* d_out, int out_size, void* d_ws, size_t ws_size,
                              hipStream_t stream) {
  const float* q  = (const float*)d_in[0];
  const float* k  = (const float*)d_in[1];
  const float* v  = (const float*)d_in[2];
  const float* eh = (const float*)d_in[3];
  const float* ew = (const float*)d_in[4];
  float* out = (float*)d_out;

  dim3 grid(BS_ * HEADS_ * (L_ / RPB_));   // 16 heads * 32 row tiles = 512
  abspos_attn_kernel<<<grid, 256, 0, stream>>>(q, k, v, eh, ew, out);
}
